// BitLinear_4887672783062
// MI455X (gfx1250) — matrix operand order repaired and checked by reading, not compiled, not run on hardware
//
#include <hip/hip_runtime.h>
#include <hip/hip_bf16.h>

// ---------------------------------------------------------------------------
// BitLinear: out = x @ absmean_quantize(W)^T
//   x: [8192, 4096] fp32, W: [4096, 4096] fp32, out: [8192, 4096] fp32
// GEMM M=8192 N=4096 K=4096 -> compute bound on MI455X (0.27 TFLOP vs 14us of
// HBM traffic). Strategy: pre-convert x->f16 and W->ternary f16 once (cheap,
// ~12us of streaming), then a TDM-fed double-buffered f16 WMMA GEMM.
// ---------------------------------------------------------------------------

typedef __attribute__((ext_vector_type(16))) _Float16 v16h;
typedef __attribute__((ext_vector_type(8)))  _Float16 v8h;
typedef __attribute__((ext_vector_type(4)))  _Float16 v4h;
typedef __attribute__((ext_vector_type(8)))  float    v8f;

typedef unsigned int u32x4 __attribute__((ext_vector_type(4)));
typedef int          i32x4 __attribute__((ext_vector_type(4)));
typedef int          i32x8 __attribute__((ext_vector_type(8)));

#define K_DIM   4096
#define N_DIM   4096
#define M_DIM   8192

// TDM GEMM tiling: block 128(M) x 256(N) x 64(K); 8 waves as 2x4, each wave
// computes a 64x64 tile = 4x4 fragments -> 16 ds_load_b128 per 16 WMMAs.
#define BM      128
#define BN      256
#define BK      64
#define LDT     72            // padded LDS row stride (f16 elements)

#if defined(__has_builtin)
#if __has_builtin(__builtin_amdgcn_tensor_load_to_lds) && \
    __has_builtin(__builtin_amdgcn_s_wait_tensorcnt)
#define HAVE_TDM 1
#endif
#endif

#if __has_include(<hip/amd_detail/amd_gfx1250_TDM.h>)
#define TDM_6ARG 1            // therock-10.0 headers -> 6-arg builtin
#endif

// ---------------- stage 1: per-block partial sums of |W| -------------------
__global__ __launch_bounds__(256)
void bitlin_abs_partial(const float* __restrict__ W, float* __restrict__ partials)
{
    __shared__ float red[256];
    const int tid = threadIdx.x;
    const size_t base4 = (size_t)blockIdx.x * 2048;   // 2048 blocks
    float s = 0.0f;
#pragma unroll
    for (int i = 0; i < 8; ++i) {
        const float4 v = *(const float4*)(W + (base4 + (size_t)i * 256 + tid) * 4);
        s += fabsf(v.x) + fabsf(v.y) + fabsf(v.z) + fabsf(v.w);
    }
    red[tid] = s;
    __syncthreads();
#pragma unroll
    for (int off = 128; off > 0; off >>= 1) {
        if (tid < off) red[tid] += red[tid + off];
        __syncthreads();
    }
    if (tid == 0) partials[blockIdx.x] = red[0];
}

// ---------------- stage 2: fold partials -> gamma (deterministic) ----------
__global__ __launch_bounds__(256)
void bitlin_abs_final(const float* __restrict__ partials, float* __restrict__ gamma)
{
    __shared__ float red[256];
    const int tid = threadIdx.x;
    float s = 0.0f;
#pragma unroll
    for (int i = 0; i < 8; ++i) s += partials[tid + i * 256];
    red[tid] = s;
    __syncthreads();
#pragma unroll
    for (int off = 128; off > 0; off >>= 1) {
        if (tid < off) red[tid] += red[tid + off];
        __syncthreads();
    }
    if (tid == 0) gamma[0] = red[0] / 16777216.0f;
}

// ---------------- stage 3a: x fp32 -> f16 ----------------------------------
__global__ __launch_bounds__(256)
void bitlin_cvt_x(const float* __restrict__ X, _Float16* __restrict__ Xh)
{
    const int tid = threadIdx.x;
    const size_t base4 = (size_t)blockIdx.x * 1024;   // 8192 blocks * 1024 float4
#pragma unroll
    for (int i = 0; i < 4; ++i) {
        const size_t e = (base4 + (size_t)i * 256 + tid) * 4;
        const float4 v = *(const float4*)(X + e);
        v4h h;
        h.x = (_Float16)v.x; h.y = (_Float16)v.y;
        h.z = (_Float16)v.z; h.w = (_Float16)v.w;
        *(v4h*)(Xh + e) = h;
    }
}

// ---------------- stage 3b: W -> ternary f16 (absmean quantize) ------------
__global__ __launch_bounds__(256)
void bitlin_quant_w(const float* __restrict__ W, const float* __restrict__ gammap,
                    _Float16* __restrict__ Wq)
{
    const float inv_g = 1.0f / (gammap[0] + 1e-5f);
    const int tid = threadIdx.x;
    const size_t base4 = (size_t)blockIdx.x * 1024;   // 4096 blocks * 1024 float4
#pragma unroll
    for (int i = 0; i < 4; ++i) {
        const size_t e = (base4 + (size_t)i * 256 + tid) * 4;
        const float4 v = *(const float4*)(W + e);
        v4h h;   // rintf == round-to-nearest-even, matches jnp.round
        h.x = (_Float16)rintf(fminf(fmaxf(v.x * inv_g, -1.0f), 1.0f));
        h.y = (_Float16)rintf(fminf(fmaxf(v.y * inv_g, -1.0f), 1.0f));
        h.z = (_Float16)rintf(fminf(fmaxf(v.z * inv_g, -1.0f), 1.0f));
        h.w = (_Float16)rintf(fminf(fmaxf(v.w * inv_g, -1.0f), 1.0f));
        *(v4h*)(Wq + e) = h;
    }
}

#if defined(HAVE_TDM)
// ---------------------------------------------------------------------------
// TDM tile fetch: 2D tile (tile_rows x 64 f16) from a row-major [rows][4096]
// f16 tensor into LDS at lds_off, with TDM padding of 4 DWORDs (8 f16) after
// every 32 DWORDs (one 64-f16 row) -> LDS row stride 72 f16 == LDT.
// ---------------------------------------------------------------------------
__device__ __forceinline__
void tdm_load_tile(unsigned lds_off, const _Float16* gptr,
                   unsigned tensor_rows, unsigned tile_rows)
{
    const unsigned long long ga = (unsigned long long)(uintptr_t)gptr;
    u32x4 g0;
    g0[0] = 1u;                                   // count=1, user descriptor
    g0[1] = lds_off;                              // LDS byte address
    g0[2] = (unsigned)ga;                         // global_addr[31:0]
    g0[3] = (unsigned)(ga >> 32) | (2u << 30);    // global_addr[56:32] | type=2

    i32x8 g1;
    // data_size=2B | pad_enable | pad_interval=32dw(code4) | pad_amount=4dw(code3)
    g1[0] = (int)((1u << 16) | (1u << 20) | (4u << 22) | (3u << 25));
    g1[1] = (int)(((unsigned)K_DIM & 0xFFFFu) << 16);              // tensor_dim0 lo16
    g1[2] = (int)(((unsigned)K_DIM >> 16) | ((tensor_rows & 0xFFFFu) << 16));
    g1[3] = (int)((tensor_rows >> 16) | ((unsigned)BK << 16));     // tile_dim0=64
    g1[4] = (int)tile_rows;                                        // tile_dim1
    g1[5] = (int)K_DIM;                                            // dim0 stride lo32
    g1[6] = 0;
    g1[7] = 0;

    const i32x4 z4 = {0, 0, 0, 0};
#if defined(TDM_6ARG)
    const i32x8 z8 = {0, 0, 0, 0, 0, 0, 0, 0};
    __builtin_amdgcn_tensor_load_to_lds(g0, g1, z4, z4, z8, 0);
#else
    __builtin_amdgcn_tensor_load_to_lds(g0, g1, z4, z4, 0);
#endif
}

// ---------------- stage 4: TDM-fed double-buffered WMMA GEMM ---------------
__global__ __launch_bounds__(256)
void bitlin_wmma_gemm_tdm(const _Float16* __restrict__ Xh,
                          const _Float16* __restrict__ Wq,
                          float* __restrict__ Out)
{
    __shared__ alignas(16) _Float16 As[2][BM * LDT];   // x tiles (f16)
    __shared__ alignas(16) _Float16 Bs[2][BN * LDT];   // Wq tiles (f16)

    const int tid  = threadIdx.x;
    const int lane = tid & 31;
    const int wv   = tid >> 5;     // 8 wave32s
    const int wm   = wv & 1;       // 2 waves along M -> 64 rows each
    const int wn   = wv >> 1;      // 4 waves along N -> 64 cols each
    const int rowF = lane & 15;
    const int hi   = lane >> 4;

    const int m0 = blockIdx.y * BM;
    const int n0 = blockIdx.x * BN;

    const unsigned aOff[2] = { (unsigned)(uintptr_t)&As[0][0],
                               (unsigned)(uintptr_t)&As[1][0] };
    const unsigned bOff[2] = { (unsigned)(uintptr_t)&Bs[0][0],
                               (unsigned)(uintptr_t)&Bs[1][0] };

    v8f acc[4][4];
#pragma unroll
    for (int i = 0; i < 4; ++i)
#pragma unroll
        for (int j = 0; j < 4; ++j) acc[i][j] = (v8f)0.0f;

    // prologue: DMA tile 0 into buffer 0
    if (wv == 0) {
        tdm_load_tile(aOff[0], Xh + (size_t)m0 * K_DIM, M_DIM, BM);
        tdm_load_tile(bOff[0], Wq + (size_t)n0 * K_DIM, N_DIM, BN);
        __builtin_amdgcn_s_wait_tensorcnt(0);
    }
    __syncthreads();

    int buf = 0;
    for (int kt = 0; kt < K_DIM; kt += BK) {
        const int nxt = buf ^ 1;
        const bool more = (kt + BK) < K_DIM;
        if (wv == 0 && more) {   // prefetch next k-tile while computing
            tdm_load_tile(aOff[nxt], Xh + (size_t)m0 * K_DIM + (kt + BK), M_DIM, BM);
            tdm_load_tile(bOff[nxt], Wq + (size_t)n0 * K_DIM + (kt + BK), N_DIM, BN);
        }

        const _Float16* __restrict__ Ab = &As[buf][0];
        const _Float16* __restrict__ Bb = &Bs[buf][0];
#pragma unroll
        for (int kk = 0; kk < BK; kk += 32) {
            v16h a[4], b[4];
#pragma unroll
            for (int fm = 0; fm < 4; ++fm) {
                // A frag (16x32): lane holds 16 contiguous K of row rowF
                const _Float16* p = Ab + (wm * 64 + fm * 16 + rowF) * LDT + kk + (hi << 4);
                const v8h lo = *(const v8h*)p;
                const v8h hh = *(const v8h*)(p + 8);
                a[fm] = __builtin_shufflevector(lo, hh, 0,1,2,3,4,5,6,7,8,9,10,11,12,13,14,15);
            }
#pragma unroll
            for (int fn = 0; fn < 4; ++fn) {
                // B frag (32x16): lane holds 16 contiguous K of column rowF
                const _Float16* p = Bb + (wn * 64 + fn * 16 + rowF) * LDT + kk + (hi << 4);
                const v8h lo = *(const v8h*)p;
                const v8h hh = *(const v8h*)(p + 8);
                b[fn] = __builtin_shufflevector(lo, hh, 0,1,2,3,4,5,6,7,8,9,10,11,12,13,14,15);
            }
#pragma unroll
            for (int fm = 0; fm < 4; ++fm)
#pragma unroll
                for (int fn = 0; fn < 4; ++fn)
                    acc[fm][fn] = __builtin_amdgcn_wmma_f32_16x16x32_f16(
                        false, a[fm], false, b[fn], (short)0, acc[fm][fn], false, false);
        }

        if (wv == 0 && more) __builtin_amdgcn_s_wait_tensorcnt(0);
        __syncthreads();
        buf = nxt;
    }

    // epilogue: C frag layout (VGPR r, lane) -> M = r + 8*hi, N = rowF
#pragma unroll
    for (int fm = 0; fm < 4; ++fm) {
#pragma unroll
        for (int fn = 0; fn < 4; ++fn) {
            const int row = m0 + wm * 64 + fm * 16 + (hi << 3);
            const int col = n0 + wn * 64 + fn * 16 + rowF;
            float* o = Out + (size_t)row * N_DIM + col;
#pragma unroll
            for (int r = 0; r < 8; ++r)
                o[(size_t)r * N_DIM] = acc[fm][fn][r];
        }
    }
}
#endif // HAVE_TDM

// ---------------- fallback: fused single-buffer WMMA GEMM ------------------
// (used when workspace is too small or TDM is unavailable)
#define FBM 128
#define FBN 128
#define FBK 64
#define FLD 72
__global__ __launch_bounds__(256)
void bitlin_wmma_gemm_fused(const float* __restrict__ X, const float* __restrict__ W,
                            const float* __restrict__ gammap, float* __restrict__ Out)
{
    __shared__ alignas(16) _Float16 As[FBM * FLD];
    __shared__ alignas(16) _Float16 Bs[FBN * FLD];

    const int tid  = threadIdx.x;
    const int lane = tid & 31;
    const int wv   = tid >> 5;
    const int wm   = wv & 3;
    const int wn   = wv >> 2;
    const int rowF = lane & 15;
    const int hi   = lane >> 4;

    const int m0 = blockIdx.y * FBM;
    const int n0 = blockIdx.x * FBN;
    const float inv_g = 1.0f / (gammap[0] + 1e-5f);

    v8f acc[2][4];
#pragma unroll
    for (int i = 0; i < 2; ++i)
#pragma unroll
        for (int j = 0; j < 4; ++j) acc[i][j] = (v8f)0.0f;

    for (int kt = 0; kt < K_DIM; kt += FBK) {
#pragma unroll
        for (int i = 0; i < 8; ++i) {
            const int idx = tid + i * 256;
            const int r = idx >> 4, c = (idx & 15) << 2;
            const float4 v = *(const float4*)(X + (size_t)(m0 + r) * K_DIM + kt + c);
            v4h h;
            h.x = (_Float16)v.x; h.y = (_Float16)v.y;
            h.z = (_Float16)v.z; h.w = (_Float16)v.w;
            *(v4h*)(As + r * FLD + c) = h;
        }
#pragma unroll
        for (int i = 0; i < 8; ++i) {
            const int idx = tid + i * 256;
            const int r = idx >> 4, c = (idx & 15) << 2;
            const float4 v = *(const float4*)(W + (size_t)(n0 + r) * K_DIM + kt + c);
            v4h h;
            h.x = (_Float16)rintf(fminf(fmaxf(v.x * inv_g, -1.0f), 1.0f));
            h.y = (_Float16)rintf(fminf(fmaxf(v.y * inv_g, -1.0f), 1.0f));
            h.z = (_Float16)rintf(fminf(fmaxf(v.z * inv_g, -1.0f), 1.0f));
            h.w = (_Float16)rintf(fminf(fmaxf(v.w * inv_g, -1.0f), 1.0f));
            *(v4h*)(Bs + r * FLD + c) = h;
        }
        __syncthreads();

#pragma unroll
        for (int kk = 0; kk < FBK; kk += 32) {
            v16h a[2], b[4];
#pragma unroll
            for (int fm = 0; fm < 2; ++fm) {
                const _Float16* p = As + (wm * 32 + fm * 16 + rowF) * FLD + kk + (hi << 4);
                const v8h lo = *(const v8h*)p;
                const v8h hh = *(const v8h*)(p + 8);
                a[fm] = __builtin_shufflevector(lo, hh, 0,1,2,3,4,5,6,7,8,9,10,11,12,13,14,15);
            }
#pragma unroll
            for (int fn = 0; fn < 4; ++fn) {
                const _Float16* p = Bs + (wn * 64 + fn * 16 + rowF) * FLD + kk + (hi << 4);
                const v8h lo = *(const v8h*)p;
                const v8h hh = *(const v8h*)(p + 8);
                b[fn] = __builtin_shufflevector(lo, hh, 0,1,2,3,4,5,6,7,8,9,10,11,12,13,14,15);
            }
#pragma unroll
            for (int fm = 0; fm < 2; ++fm)
#pragma unroll
                for (int fn = 0; fn < 4; ++fn)
                    acc[fm][fn] = __builtin_amdgcn_wmma_f32_16x16x32_f16(
                        false, a[fm], false, b[fn], (short)0, acc[fm][fn], false, false);
        }
        __syncthreads();
    }

#pragma unroll
    for (int fm = 0; fm < 2; ++fm) {
#pragma unroll
        for (int fn = 0; fn < 4; ++fn) {
            const int row = m0 + wm * 32 + fm * 16 + (hi << 3);
            const int col = n0 + wn * 64 + fn * 16 + rowF;
            float* o = Out + (size_t)row * N_DIM + col;
#pragma unroll
            for (int r = 0; r < 8; ++r)
                o[(size_t)r * N_DIM] = acc[fm][fn][r];
        }
    }
}

// ---------------------------------------------------------------------------
extern "C" void kernel_launch(void* const* d_in, const int* in_sizes, int n_in,
                              void* d_out, int out_size, void* d_ws, size_t ws_size,
                              hipStream_t stream)
{
    const float* X = (const float*)d_in[0];   // [4,2048,4096] fp32
    const float* W = (const float*)d_in[1];   // [4096,4096]   fp32
    float* out = (float*)d_out;

    float* gamma    = (float*)d_ws;           // ws[0]
    float* partials = gamma + 64;             // 2048 floats at +256B

    bitlin_abs_partial<<<2048, 256, 0, stream>>>(W, partials);
    bitlin_abs_final  <<<1,    256, 0, stream>>>(partials, gamma);

#if defined(HAVE_TDM)
    const size_t XH_OFF = 65536;
    const size_t WQ_OFF = XH_OFF + sizeof(_Float16) * (size_t)M_DIM * K_DIM;
    const size_t need   = WQ_OFF + sizeof(_Float16) * (size_t)N_DIM * K_DIM;
    if (ws_size >= need) {
        _Float16* Xh = (_Float16*)((char*)d_ws + XH_OFF);
        _Float16* Wq = (_Float16*)((char*)d_ws + WQ_OFF);
        bitlin_cvt_x  <<<8192, 256, 0, stream>>>(X, Xh);
        bitlin_quant_w<<<4096, 256, 0, stream>>>(W, gamma, Wq);
        dim3 grid(N_DIM / BN, M_DIM / BM);    // (16, 64)
        bitlin_wmma_gemm_tdm<<<grid, 256, 0, stream>>>(Xh, Wq, out);
        return;
    }
#endif
    dim3 fgrid(N_DIM / FBN, M_DIM / FBM);     // (32, 64)
    bitlin_wmma_gemm_fused<<<fgrid, 256, 0, stream>>>(X, W, gamma, out);
}
